// Completion_76725295775949
// MI455X (gfx1250) — matrix operand order repaired and checked by reading, not compiled, not run on hardware
//
#include <hip/hip_runtime.h>
#include <hip/hip_bf16.h>

typedef __attribute__((ext_vector_type(16))) _Float16 v16h;
typedef __attribute__((ext_vector_type(8)))  _Float16 v8h;
typedef __attribute__((ext_vector_type(8)))  float    v8f;

#define B_   256
#define T_   64
#define IN_  96
#define H_   1024
#define OUT_ 96
#define G4H_ (4 * H_)

// ---------------------------------------------------------------------------
// math helpers (f32, fast transcendental path -> v_exp_f32)
// ---------------------------------------------------------------------------
__device__ __forceinline__ float sigf(float x) {
    return 1.0f / (1.0f + __expf(-x));
}
__device__ __forceinline__ float tanh_fast(float x) {
    x = fminf(fmaxf(x, -15.0f), 15.0f);
    float e = __expf(2.0f * x);
    return (e - 1.0f) / (e + 1.0f);
}

// ---------------------------------------------------------------------------
// WMMA fragment loaders for V_WMMA_F32_16X16X32_F16 (wave32)
// A: 16x32 (MxK) f16, row-major source with leading dim `ld` (halfs).
//   lane m = lane&15 ; holds K = k0..k0+15, k0 = (lane>=16)?16:0
// B: 32x16 (KxN) f16; source is W stored [N][K] row-major (i.e. W[4H][K]).
//   lane n = lane&15 ; holds K = kb..kb+15, kb = (lane>=16)?16:0
// ---------------------------------------------------------------------------
__device__ __forceinline__ v16h load_a_frag(const _Float16* __restrict__ base,
                                            int ld, int lane) {
    const int m  = lane & 15;
    const int k0 = (lane >> 4) << 4;          // 0 or 16
    const _Float16* p = base + m * ld + k0;
    v8h lo = *(const v8h*)(p);                // K = k0 .. k0+7
    v8h hi = *(const v8h*)(p + 8);            // K = k0+8 .. k0+15
    v16h r;
#pragma unroll
    for (int e = 0; e < 8; ++e) { r[e] = lo[e]; r[e + 8] = hi[e]; }
    return r;
}

__device__ __forceinline__ v16h load_b_frag(const _Float16* __restrict__ base,
                                            int ld, int lane) {
    const int n  = lane & 15;
    const int kb = (lane >> 4) << 4;          // 0 or 16
    const _Float16* p = base + n * ld + kb;
    v8h lo = *(const v8h*)(p);
    v8h hi = *(const v8h*)(p + 8);
    v16h r;
#pragma unroll
    for (int e = 0; e < 8; ++e) { r[e] = lo[e]; r[e + 8] = hi[e]; }
    return r;
}

#define WMMA_F16(A, Bf, C) \
    __builtin_amdgcn_wmma_f32_16x16x32_f16(false, (A), false, (Bf), (short)0, (C), false, false)

// ---------------------------------------------------------------------------
// Fused LSTM cell step: gates = x@Wih^T + b_ih + h@Whh^T + b_hh, then
// elementwise cell update.
// One wave computes a 32x16 tile of [B,H]: 2 M-subtiles x 4 gates = 8
// accumulators; each W fragment load feeds 2 WMMAs (halves L2 weight traffic).
// Block = 8 waves = 2 B-supertiles(32 rows) x 4 H-tiles.
// Grid = 64 blocks covering 8 B-supertiles x 64 H-tiles.
// ---------------------------------------------------------------------------
__global__ void __launch_bounds__(256)
lstm_step_kernel(const _Float16* __restrict__ x, int ldx, int Kx,
                 const _Float16* __restrict__ Wih,    // [4H, Kx] row-major
                 const _Float16* __restrict__ hprev,  // [B, H]
                 const _Float16* __restrict__ Whh,    // [4H, H] row-major
                 const float* __restrict__ b_ih,      // [4H]
                 const float* __restrict__ b_hh,      // [4H]
                 float* __restrict__ c,               // [B, H] in/out
                 _Float16* __restrict__ hout)         // [B, H]
{
    const int lane = threadIdx.x & 31;
    const int wave = threadIdx.x >> 5;
    const int hgrp = blockIdx.x & 15;     // 16 groups * 4 H-tiles = 64
    const int bgrp = blockIdx.x >> 4;     // 4 groups * 2 B-supertiles = 8
    const int j0 = (hgrp * 4 + (wave & 3)) * 16;   // H column base
    const int b0 = (bgrp * 2 + (wave >> 2)) * 32;  // B row base (32-row tile)

    v8f ai0 = {}, af0 = {}, ag0 = {}, ao0 = {};
    v8f ai1 = {}, af1 = {}, ag1 = {}, ao1 = {};

    // ---- phase 1: x @ Wih^T ----
    for (int k = 0; k < Kx; k += 32) {
        v16h a0 = load_a_frag(x + (size_t)b0 * ldx + k, ldx, lane);
        v16h a1 = load_a_frag(x + (size_t)(b0 + 16) * ldx + k, ldx, lane);
        v16h wi = load_b_frag(Wih + (size_t)(0 * H_ + j0) * Kx + k, Kx, lane);
        v16h wf = load_b_frag(Wih + (size_t)(1 * H_ + j0) * Kx + k, Kx, lane);
        v16h wg = load_b_frag(Wih + (size_t)(2 * H_ + j0) * Kx + k, Kx, lane);
        v16h wo = load_b_frag(Wih + (size_t)(3 * H_ + j0) * Kx + k, Kx, lane);
        ai0 = WMMA_F16(a0, wi, ai0);  ai1 = WMMA_F16(a1, wi, ai1);
        af0 = WMMA_F16(a0, wf, af0);  af1 = WMMA_F16(a1, wf, af1);
        ag0 = WMMA_F16(a0, wg, ag0);  ag1 = WMMA_F16(a1, wg, ag1);
        ao0 = WMMA_F16(a0, wo, ao0);  ao1 = WMMA_F16(a1, wo, ao1);
    }

    // ---- phase 2: h @ Whh^T ----
    for (int k = 0; k < H_; k += 32) {
        v16h a0 = load_a_frag(hprev + (size_t)b0 * H_ + k, H_, lane);
        v16h a1 = load_a_frag(hprev + (size_t)(b0 + 16) * H_ + k, H_, lane);
        v16h wi = load_b_frag(Whh + (size_t)(0 * H_ + j0) * H_ + k, H_, lane);
        v16h wf = load_b_frag(Whh + (size_t)(1 * H_ + j0) * H_ + k, H_, lane);
        v16h wg = load_b_frag(Whh + (size_t)(2 * H_ + j0) * H_ + k, H_, lane);
        v16h wo = load_b_frag(Whh + (size_t)(3 * H_ + j0) * H_ + k, H_, lane);
        ai0 = WMMA_F16(a0, wi, ai0);  ai1 = WMMA_F16(a1, wi, ai1);
        af0 = WMMA_F16(a0, wf, af0);  af1 = WMMA_F16(a1, wf, af1);
        ag0 = WMMA_F16(a0, wg, ag0);  ag1 = WMMA_F16(a1, wg, ag1);
        ao0 = WMMA_F16(a0, wo, ao0);  ao1 = WMMA_F16(a1, wo, ao1);
    }

    // ---- fused gate elementwise + cell update ----
    const int n   = lane & 15;
    const int col = j0 + n;
    const float bi = b_ih[0 * H_ + col] + b_hh[0 * H_ + col];
    const float bf = b_ih[1 * H_ + col] + b_hh[1 * H_ + col];
    const float bg = b_ih[2 * H_ + col] + b_hh[2 * H_ + col];
    const float bo = b_ih[3 * H_ + col] + b_hh[3 * H_ + col];
    const int mofs = (lane >> 4) << 3;     // C/D layout: M = r + (lane>=16)*8

#pragma unroll
    for (int r = 0; r < 8; ++r) {
        // M-subtile 0 (rows b0 .. b0+15)
        {
            const int row = b0 + r + mofs;
            const size_t idx = (size_t)row * H_ + col;
            float iv = sigf(ai0[r] + bi);
            float fv = sigf(af0[r] + bf);
            float gv = tanh_fast(ag0[r] + bg);
            float ov = sigf(ao0[r] + bo);
            float cn = fv * c[idx] + iv * gv;
            c[idx]   = cn;
            hout[idx] = (_Float16)(ov * tanh_fast(cn));
        }
        // M-subtile 1 (rows b0+16 .. b0+31)
        {
            const int row = b0 + 16 + r + mofs;
            const size_t idx = (size_t)row * H_ + col;
            float iv = sigf(ai1[r] + bi);
            float fv = sigf(af1[r] + bf);
            float gv = tanh_fast(ag1[r] + bg);
            float ov = sigf(ao1[r] + bo);
            float cn = fv * c[idx] + iv * gv;
            c[idx]   = cn;
            hout[idx] = (_Float16)(ov * tanh_fast(cn));
        }
    }
}

// ---------------------------------------------------------------------------
// Output projection: out[b,t,:] = sigmoid(h1 @ Wout^T + b_out)
// One wave per 16x16 tile; grid = 16 B-tiles * 6 OUT-tiles = 96 blocks.
// ---------------------------------------------------------------------------
__global__ void __launch_bounds__(32)
out_proj_kernel(const _Float16* __restrict__ h1,     // [B, H]
                const _Float16* __restrict__ Wout,   // [OUT, H] row-major
                const float* __restrict__ bout,      // [OUT]
                float* __restrict__ out, int t)      // out: [B, T, OUT]
{
    const int lane = threadIdx.x & 31;
    const int ot = blockIdx.x % 6;
    const int bt = blockIdx.x / 6;
    const int o0 = ot * 16;
    const int b0 = bt * 16;

    v8f acc = {};
    for (int k = 0; k < H_; k += 32) {
        v16h a = load_a_frag(h1 + (size_t)b0 * H_ + k, H_, lane);
        v16h w = load_b_frag(Wout + (size_t)o0 * H_ + k, H_, lane);
        acc = WMMA_F16(a, w, acc);
    }

    const int n = lane & 15;
    const int col = o0 + n;
    const float bb = bout[col];
    const int mofs = (lane >> 4) << 3;
#pragma unroll
    for (int r = 0; r < 8; ++r) {
        const int row = b0 + r + mofs;
        out[(size_t)row * (T_ * OUT_) + (size_t)t * OUT_ + col] = sigf(acc[r] + bb);
    }
}

// ---------------------------------------------------------------------------
// one-shot conversion kernels
// ---------------------------------------------------------------------------
__global__ void f32_to_f16_kernel(const float* __restrict__ src,
                                  _Float16* __restrict__ dst, int n) {
    int i = blockIdx.x * blockDim.x + threadIdx.x;
    if (i < n) dst[i] = (_Float16)src[i];
}

extern "C" void kernel_launch(void* const* d_in, const int* in_sizes, int n_in,
                              void* d_out, int out_size, void* d_ws, size_t ws_size,
                              hipStream_t stream) {
    (void)in_sizes; (void)n_in; (void)out_size; (void)ws_size;

    const float* inputs  = (const float*)d_in[0];   // [B, T, IN]
    const float* hiddens = (const float*)d_in[1];   // [L, B, H]
    const float* cells   = (const float*)d_in[2];   // [L, B, H]
    const float* W_ih0   = (const float*)d_in[3];   // [4H, IN]
    const float* W_hh0   = (const float*)d_in[4];   // [4H, H]
    const float* b_ih0   = (const float*)d_in[5];
    const float* b_hh0   = (const float*)d_in[6];
    const float* W_ih1   = (const float*)d_in[7];   // [4H, H]
    const float* W_hh1   = (const float*)d_in[8];   // [4H, H]
    const float* b_ih1   = (const float*)d_in[9];
    const float* b_hh1   = (const float*)d_in[10];
    const float* W_out   = (const float*)d_in[11];  // [OUT, H]
    const float* b_out   = (const float*)d_in[12];
    float* out = (float*)d_out;

    // ---- workspace carve-out (all 256B aligned) ----
    char* ws = (char*)d_ws;
    size_t off = 0;
    auto carve = [&](size_t bytes) -> void* {
        void* p = ws + off;
        off = (off + bytes + 255) & ~(size_t)255;
        return p;
    };
    _Float16* xh   = (_Float16*)carve((size_t)B_ * T_ * IN_ * 2);
    _Float16* wih0 = (_Float16*)carve((size_t)G4H_ * IN_ * 2);
    _Float16* whh0 = (_Float16*)carve((size_t)G4H_ * H_ * 2);
    _Float16* wih1 = (_Float16*)carve((size_t)G4H_ * H_ * 2);
    _Float16* whh1 = (_Float16*)carve((size_t)G4H_ * H_ * 2);
    _Float16* wout = (_Float16*)carve((size_t)OUT_ * H_ * 2);
    _Float16* h0b[2] = { (_Float16*)carve((size_t)B_ * H_ * 2),
                         (_Float16*)carve((size_t)B_ * H_ * 2) };
    _Float16* h1b[2] = { (_Float16*)carve((size_t)B_ * H_ * 2),
                         (_Float16*)carve((size_t)B_ * H_ * 2) };
    float* c0 = (float*)carve((size_t)B_ * H_ * 4);
    float* c1 = (float*)carve((size_t)B_ * H_ * 4);

    // ---- one-time conversions (weights become L2-resident f16) ----
    auto cvt = [&](const float* s, _Float16* d, int n) {
        f32_to_f16_kernel<<<(n + 255) / 256, 256, 0, stream>>>(s, d, n);
    };
    cvt(inputs, xh, B_ * T_ * IN_);
    cvt(W_ih0, wih0, G4H_ * IN_);
    cvt(W_hh0, whh0, G4H_ * H_);
    cvt(W_ih1, wih1, G4H_ * H_);
    cvt(W_hh1, whh1, G4H_ * H_);
    cvt(W_out, wout, OUT_ * H_);
    cvt(hiddens,            h0b[0], B_ * H_);
    cvt(hiddens + B_ * H_,  h1b[0], B_ * H_);
    hipMemcpyAsync(c0, cells,            (size_t)B_ * H_ * 4, hipMemcpyDeviceToDevice, stream);
    hipMemcpyAsync(c1, cells + B_ * H_,  (size_t)B_ * H_ * 4, hipMemcpyDeviceToDevice, stream);

    // ---- recurrence: per-step kernels give grid-wide ordering on stream ----
    for (int t = 0; t < T_; ++t) {
        const int cur = t & 1, nxt = (t + 1) & 1;
        // layer 0: x_t is a strided [B, IN] view of inputs (ld = T*IN)
        lstm_step_kernel<<<64, 256, 0, stream>>>(
            xh + (size_t)t * IN_, T_ * IN_, IN_,
            wih0, h0b[cur], whh0, b_ih0, b_hh0, c0, h0b[nxt]);
        // layer 1: x = freshly written h0
        lstm_step_kernel<<<64, 256, 0, stream>>>(
            h0b[nxt], H_, H_,
            wih1, h1b[cur], whh1, b_ih1, b_hh1, c1, h1b[nxt]);
        // projection + sigmoid -> out[:, t, :]
        out_proj_kernel<<<96, 32, 0, stream>>>(h1b[nxt], wout, b_out, out, t);
    }
}
